// PMBNNAlwaysPlasticV2_39015482917363
// MI455X (gfx1250) — compile-verified, matrix operand order checked by reading, not run on hardware
//
#include <hip/hip_runtime.h>
#include <stdint.h>

// ---------------- problem constants (from reference) ----------------
#define B_SZ   16384
#define D_IN   784
#define K_PAD  800            // 25 * 32, zero-padded K
#define D_HID  256
#define D_LAT  8
#define N_C    64
#define CH_N   64
#define N_CLS  10
#define T_STEPS 5
#define PM_STEPS 4

#define DT_C   0.1f
#define EPS_C  1e-4f
#define TAU_C  0.9f
#define MG_C   0.1f
#define EI_C   0.06f
#define PLR_C  0.001f

// output layout: logits [B,10] | z [B,8] | h [B,64]
#define OUT_Z_OFF  (B_SZ * N_CLS)
#define OUT_H_OFF  (B_SZ * (N_CLS + D_LAT))

// ---------------- workspace layout (bytes) ----------------
#define W1T_OFF  0u                                               // [256][800] bf16
#define Z_OFF    (W1T_OFF + (size_t)D_HID * K_PAD * 2)            // 409600
#define H_OFF    (Z_OFF   + (size_t)B_SZ * D_LAT * 4)
#define MU_OFF   (H_OFF   + (size_t)B_SZ * CH_N * 4)
#define C_OFF    (MU_OFF  + 256u)
#define S1_OFF   (C_OFF   + (size_t)N_C * D_LAT * 4)
#define S3_OFF   (S1_OFF  + 256u)
#define S2_OFF   (S3_OFF  + 256u)

// ---------------- WMMA types ----------------
typedef __attribute__((ext_vector_type(16))) __bf16 v16bf;
typedef __attribute__((ext_vector_type(8)))  float  v8f;
union BF16Frag { v16bf v; unsigned u[8]; };

__device__ __forceinline__ uint16_t f2bf(float f) {
    unsigned u = __float_as_uint(f);
    unsigned r = u + 0x7FFFu + ((u >> 16) & 1u);   // RNE
    return (uint16_t)(r >> 16);
}

// ---------------- kernel 1: W1 f32 -> bf16, transposed [n][k], K zero-padded ----------------
__global__ void prep_w1t_kernel(const float* __restrict__ W1, uint16_t* __restrict__ w1t) {
    int i = blockIdx.x * 256 + threadIdx.x;      // over 256*800
    if (i < D_HID * K_PAD) {
        const int n = i / K_PAD, k = i % K_PAD;
        w1t[i] = (k < D_IN) ? f2bf(W1[(size_t)k * D_HID + n]) : (uint16_t)0;
    }
}

// ---------------- kernel 2: fused encoder ----------------
// H = tanh(x @ W1 + b1) kept in VGPRs; z = H @ W2 + b2 fused in epilogue; h state zeroed.
// Workgroup: 256 threads = 8 waves, tile 64 rows x 256 cols; waves 4(M) x 2(N),
// each wave 16 rows x 128 cols = 8 bf16 WMMA accumulator tiles.
// B tile staged LDS via CDNA5 async-to-LDS (ASYNCcnt); A converted f32->bf16 in flight.
__global__ __launch_bounds__(256)
void enc_fused_kernel(const float* __restrict__ x, const uint16_t* __restrict__ w1t,
                      const float* __restrict__ b1, const float* __restrict__ W2,
                      const float* __restrict__ b2,
                      float* __restrict__ z_ws, float* __restrict__ h_ws) {
    __shared__ uint16_t Als[64 * 32];     // [row][k] bf16
    __shared__ uint16_t Bls[256 * 32];    // [n][k]   bf16
    __shared__ float    z_lds[64 * 8];

    const int tid  = threadIdx.x;
    const int lane = tid & 31;
    const int wave = tid >> 5;
    const int wv_m = wave >> 1;           // 0..3
    const int wv_n = wave & 1;            // 0..1
    const int m0   = blockIdx.x * 64;
    const int hrow = lane & 15;
    const int kb   = (lane & 16) ? 8 : 0; // K base per lane-half (ISA 16-bit layout)

    for (int i = tid; i < 64 * 8; i += 256) z_lds[i] = 0.f;

    v8f acc[8];
#pragma unroll
    for (int t = 0; t < 8; ++t) acc[t] = (v8f){0.f,0.f,0.f,0.f,0.f,0.f,0.f,0.f};

    const int ar = tid >> 2;              // A stage: row 0..63
    const int ac = (tid & 3) * 8;         // A stage: col seg 0/8/16/24

    auto kbody = [&](int k0, bool tail) {
        // ---- async copy B tile: row n = tid, 64 contiguous bytes ----
        {
            const uint64_t ga = (uint64_t)(uintptr_t)(w1t + (size_t)tid * K_PAD + k0);
            const unsigned lo = (unsigned)(uintptr_t)(&Bls[tid * 32]);
            asm volatile("global_load_async_to_lds_b128 %0, %1, off"           :: "v"(lo), "v"(ga) : "memory");
            asm volatile("global_load_async_to_lds_b128 %0, %1, off offset:16" :: "v"(lo), "v"(ga) : "memory");
            asm volatile("global_load_async_to_lds_b128 %0, %1, off offset:32" :: "v"(lo), "v"(ga) : "memory");
            asm volatile("global_load_async_to_lds_b128 %0, %1, off offset:48" :: "v"(lo), "v"(ga) : "memory");
        }
        // ---- stage A: 64x32 f32 -> bf16 (8 elems/thread) ----
        {
            const float* src = x + (size_t)(m0 + ar) * D_IN + k0 + ac;
            uint16_t* dst = &Als[ar * 32 + ac];
            if (!tail || ac < 16) {       // tail chunk: only segments 0/8 are in-bounds
                const float4 f0 = *(const float4*)(src);
                const float4 f1 = *(const float4*)(src + 4);
                dst[0]=f2bf(f0.x); dst[1]=f2bf(f0.y); dst[2]=f2bf(f0.z); dst[3]=f2bf(f0.w);
                dst[4]=f2bf(f1.x); dst[5]=f2bf(f1.y); dst[6]=f2bf(f1.z); dst[7]=f2bf(f1.w);
                if (!tail) __builtin_prefetch(src + 32, 0, 3);
            } else {
#pragma unroll
                for (int j = 0; j < 8; ++j) dst[j] = 0;
            }
        }
        asm volatile("s_wait_asynccnt 0x0" ::: "memory");
        __syncthreads();

        // ---- A fragment (ISA 16-bit A layout: pairs at K = kb+2j, 16+kb+2j) ----
        BF16Frag af;
        {
            const uint16_t* ap = &Als[(wv_m * 16 + hrow) * 32];
#pragma unroll
            for (int j = 0; j < 4; ++j) {
                af.u[j]     = *(const unsigned*)(ap + kb + 2 * j);
                af.u[4 + j] = *(const unsigned*)(ap + 16 + kb + 2 * j);
            }
        }
        // ---- preload all 8 B fragments, then back-to-back WMMA ----
        BF16Frag bfr[8];
#pragma unroll
        for (int t = 0; t < 8; ++t) {
            const uint16_t* bpp = &Bls[(wv_n * 128 + t * 16 + hrow) * 32];
#pragma unroll
            for (int j = 0; j < 4; ++j) {
                bfr[t].u[j]     = *(const unsigned*)(bpp + kb + 2 * j);
                bfr[t].u[4 + j] = *(const unsigned*)(bpp + 16 + kb + 2 * j);
            }
        }
#pragma unroll
        for (int t = 0; t < 8; ++t) {
            acc[t] = __builtin_amdgcn_wmma_f32_16x16x32_bf16(
                         false, af.v, false, bfr[t].v, (short)0, acc[t], false, false);
        }
        __syncthreads();
    };

    for (int kc = 0; kc < 24; ++kc) kbody(kc * 32, false);  // full chunks, no guards
    kbody(768, true);                                        // guarded tail chunk

    // ---- epilogue: tanh(H) and fused z = H @ W2 partial sums ----
    float pz[8][8];
#pragma unroll
    for (int r = 0; r < 8; ++r)
#pragma unroll
        for (int d = 0; d < D_LAT; ++d) pz[r][d] = 0.f;

#pragma unroll
    for (int t = 0; t < 8; ++t) {
        const int ncol = wv_n * 128 + t * 16 + hrow;
        const float bias = b1[ncol];
        const float4 w2a = *(const float4*)(W2 + ncol * D_LAT);
        const float4 w2b = *(const float4*)(W2 + ncol * D_LAT + 4);
        const float w2r[8] = {w2a.x, w2a.y, w2a.z, w2a.w, w2b.x, w2b.y, w2b.z, w2b.w};
#pragma unroll
        for (int r = 0; r < 8; ++r) {
            const float hv = tanhf(acc[t][r] + bias);
#pragma unroll
            for (int d = 0; d < D_LAT; ++d) pz[r][d] += hv * w2r[d];
        }
    }
    // reduce over the 16 lanes of each half-wave (different ncol, same rows)
#pragma unroll
    for (int m = 8; m >= 1; m >>= 1)
#pragma unroll
        for (int r = 0; r < 8; ++r)
#pragma unroll
            for (int d = 0; d < D_LAT; ++d) pz[r][d] += __shfl_xor(pz[r][d], m);

    const int rloc = wv_m * 16 + ((lane & 16) ? 8 : 0);  // local row base 0..63
    if (hrow == 0) {
#pragma unroll
        for (int r = 0; r < 8; ++r)
#pragma unroll
            for (int d = 0; d < D_LAT; ++d)
                atomicAdd(&z_lds[(rloc + r) * D_LAT + d], pz[r][d]);
    }
    __syncthreads();

    for (int i = tid; i < 64 * D_LAT; i += 256)
        z_ws[(size_t)(m0 + i / D_LAT) * D_LAT + (i & 7)] = z_lds[i] + b2[i & 7];
    // zero membrane state for these 64 rows (contiguous 4096 floats)
    float4* hz = (float4*)(h_ws + (size_t)m0 * CH_N);
    for (int i = tid; i < 64 * CH_N / 4; i += 256) hz[i] = (float4){0.f,0.f,0.f,0.f};
}

// ---------------- kernel 3: init mu/c/accumulators ----------------
__global__ void pm_init_kernel(const float* __restrict__ mu, const float* __restrict__ c,
                               float* mu_ws, float* c_ws, float* S1, float* S2, float* S3) {
    const int tid = threadIdx.x;
    if (tid < N_C) { mu_ws[tid] = mu[tid]; S1[tid] = 0.f; S3[tid] = 0.f; }
    for (int i = tid; i < N_C * D_LAT; i += 256) { c_ws[i] = c[i]; S2[i] = 0.f; }
}

// ---------------- kernel 4: one PM-field time step ----------------
__global__ __launch_bounds__(256)
void pm_step_kernel(float* __restrict__ z_ws, float* __restrict__ h_ws,
                    const float* __restrict__ mu_ws, const float* __restrict__ c_ws,
                    const float* __restrict__ Wp, const float* __restrict__ bp,
                    const float* __restrict__ Wr, const float* __restrict__ br,
                    float* __restrict__ S1, float* __restrict__ S2, float* __restrict__ S3,
                    float* __restrict__ out, int last) {
    __shared__ float c_s[N_C * D_LAT];
    __shared__ float Wp_s[D_LAT * CH_N];
    __shared__ float mu_s[N_C], bp_s[N_C];
    __shared__ float r1s[N_C], r3s[N_C], r2sh[N_C * D_LAT];

    const int tid = threadIdx.x;
    for (int i = tid; i < N_C * D_LAT; i += 256) { c_s[i] = c_ws[i]; Wp_s[i] = Wp[i]; r2sh[i] = 0.f; }
    if (tid < N_C) { mu_s[tid] = mu_ws[tid]; bp_s[tid] = bp[tid]; r1s[tid] = 0.f; r3s[tid] = 0.f; }
    __syncthreads();

    const int b = blockIdx.x * 256 + tid;
    float z[D_LAT];
#pragma unroll
    for (int d = 0; d < D_LAT; ++d) z[d] = z_ws[b * D_LAT + d];

    // PM evolve: 4 gravitational sub-steps
#pragma unroll
    for (int it = 0; it < PM_STEPS; ++it) {
        float accd[D_LAT] = {0.f,0.f,0.f,0.f,0.f,0.f,0.f,0.f};
        for (int n = 0; n < N_C; ++n) {
            float r2 = EPS_C, dd[D_LAT];
#pragma unroll
            for (int d = 0; d < D_LAT; ++d) { dd[d] = c_s[n * D_LAT + d] - z[d]; r2 += dd[d] * dd[d]; }
            const float s = mu_s[n] / r2;
#pragma unroll
            for (int d = 0; d < D_LAT; ++d) accd[d] += s * dd[d];
        }
#pragma unroll
        for (int d = 0; d < D_LAT; ++d) z[d] += DT_C * accd[d];
    }

    // membrane + lateral EI
    float hf[CH_N];
    float hsum = 0.f;
    for (int ch = 0; ch < CH_N; ++ch) {
        float a = bp_s[ch];
#pragma unroll
        for (int d = 0; d < D_LAT; ++d) a += z[d] * Wp_s[d * CH_N + ch];
        const float hn = TAU_C * h_ws[(size_t)b * CH_N + ch] + MG_C * tanhf(a);
        hf[ch] = hn; hsum += hn;
    }
    const float hm = hsum * (1.f / CH_N);
    float psum = 0.f;
    for (int ch = 0; ch < CH_N; ++ch) { hf[ch] += EI_C * (hf[ch] - hm); psum += hf[ch]; }
    const float post = psum * (1.f / CH_N);

    // carry state (and outputs on last step)
#pragma unroll
    for (int d = 0; d < D_LAT; ++d) z_ws[b * D_LAT + d] = z[d];
    for (int ch = 0; ch < CH_N; ++ch) h_ws[(size_t)b * CH_N + ch] = hf[ch];
    if (last) {
#pragma unroll
        for (int d = 0; d < D_LAT; ++d) out[OUT_Z_OFF + b * D_LAT + d] = z[d];
        for (int ch = 0; ch < CH_N; ++ch) out[OUT_H_OFF + (size_t)b * CH_N + ch] = hf[ch];
        for (int k = 0; k < N_CLS; ++k) {
            float l = br[k];
            for (int ch = 0; ch < CH_N; ++ch) l += hf[ch] * Wr[ch * N_CLS + k];
            out[b * N_CLS + k] = l;
        }
    }

    // plasticity partials: wave32 shfl_xor reduction -> LDS atomics -> global atomics
    const int lane = tid & 31;
    for (int n = 0; n < N_C; ++n) {
        float r2 = 0.f;
#pragma unroll
        for (int d = 0; d < D_LAT; ++d) { const float dd = c_s[n * D_LAT + d] - z[d]; r2 += dd * dd; }
        const float act = __expf(-r2);
        float v0 = act, v1 = act * post, vz[D_LAT];
#pragma unroll
        for (int d = 0; d < D_LAT; ++d) vz[d] = act * z[d];
#pragma unroll
        for (int m = 16; m >= 1; m >>= 1) {
            v0 += __shfl_xor(v0, m);
            v1 += __shfl_xor(v1, m);
#pragma unroll
            for (int d = 0; d < D_LAT; ++d) vz[d] += __shfl_xor(vz[d], m);
        }
        if (lane == 0) {
            atomicAdd(&r1s[n], v0);
            atomicAdd(&r3s[n], v1);
#pragma unroll
            for (int d = 0; d < D_LAT; ++d) atomicAdd(&r2sh[n * D_LAT + d], vz[d]);
        }
    }
    __syncthreads();
    if (tid < N_C) { atomicAdd(&S1[tid], r1s[tid]); atomicAdd(&S3[tid], r3s[tid]); }
    for (int i = tid; i < N_C * D_LAT; i += 256) atomicAdd(&S2[i], r2sh[i]);
}

// ---------------- kernel 5: finalize mu/c update, re-zero accumulators ----------------
__global__ void pm_fin_kernel(float* mu_ws, float* c_ws, float* S1, float* S2, float* S3) {
    const int tid = threadIdx.x;
    float wsum = 0.f, s3 = 0.f, s2[D_LAT];
    if (tid < N_C) {
        wsum = S1[tid]; s3 = S3[tid];
#pragma unroll
        for (int d = 0; d < D_LAT; ++d) s2[d] = S2[tid * D_LAT + d];
    }
    __syncthreads();
    if (tid < N_C) {
        mu_ws[tid] += PLR_C * (s3 * (1.f / (float)B_SZ));
        const float inv = 1.f / (wsum + EPS_C);
        const float frac = wsum * inv;
#pragma unroll
        for (int d = 0; d < D_LAT; ++d)
            c_ws[tid * D_LAT + d] += PLR_C * (s2[d] * inv - c_ws[tid * D_LAT + d] * frac);
        S1[tid] = 0.f; S3[tid] = 0.f;
    }
    for (int i = tid; i < N_C * D_LAT; i += 256) S2[i] = 0.f;
}

// ---------------- host launcher ----------------
extern "C" void kernel_launch(void* const* d_in, const int* in_sizes, int n_in,
                              void* d_out, int out_size, void* d_ws, size_t ws_size,
                              hipStream_t stream) {
    const float* x  = (const float*)d_in[0];
    const float* W1 = (const float*)d_in[1];
    const float* b1 = (const float*)d_in[2];
    const float* W2 = (const float*)d_in[3];
    const float* b2 = (const float*)d_in[4];
    const float* mu = (const float*)d_in[5];
    const float* c  = (const float*)d_in[6];
    const float* Wp = (const float*)d_in[7];
    const float* bp = (const float*)d_in[8];
    const float* Wr = (const float*)d_in[9];
    const float* br = (const float*)d_in[10];
    // d_in[11] is T (== 5), hardcoded as T_STEPS.

    char* ws = (char*)d_ws;
    uint16_t* w1t  = (uint16_t*)(ws + W1T_OFF);
    float* z_ws  = (float*)(ws + Z_OFF);
    float* h_ws  = (float*)(ws + H_OFF);
    float* mu_ws = (float*)(ws + MU_OFF);
    float* c_ws  = (float*)(ws + C_OFF);
    float* S1    = (float*)(ws + S1_OFF);
    float* S3    = (float*)(ws + S3_OFF);
    float* S2    = (float*)(ws + S2_OFF);
    float* out   = (float*)d_out;

    prep_w1t_kernel<<<(D_HID * K_PAD + 255) / 256, 256, 0, stream>>>(W1, w1t);
    enc_fused_kernel<<<B_SZ / 64, 256, 0, stream>>>(x, w1t, b1, W2, b2, z_ws, h_ws);
    pm_init_kernel<<<1, 256, 0, stream>>>(mu, c, mu_ws, c_ws, S1, S2, S3);

    for (int t = 0; t < T_STEPS; ++t) {
        pm_step_kernel<<<B_SZ / 256, 256, 0, stream>>>(z_ws, h_ws, mu_ws, c_ws,
                                                       Wp, bp, Wr, br,
                                                       S1, S2, S3, out,
                                                       (t == T_STEPS - 1) ? 1 : 0);
        if (t < T_STEPS - 1)
            pm_fin_kernel<<<1, 256, 0, stream>>>(mu_ws, c_ws, S1, S2, S3);
    }
}
